// GCAE_25048249270384
// MI455X (gfx1250) — matrix operand order repaired and checked by reading, not compiled, not run on hardware
//
#include <hip/hip_runtime.h>
#include <stdint.h>

typedef __attribute__((ext_vector_type(16))) __bf16 v16bf;
typedef __attribute__((ext_vector_type(8)))  float  v8f;

// ---------------------------------------------------------------------------
// Degree / normalization:  dinv[i] = (deg_in(i) + 1)^-0.5   (self-loop incl.)
// ---------------------------------------------------------------------------
__global__ void k_fill(float* __restrict__ p, float v, int n) {
    int i = blockIdx.x * blockDim.x + threadIdx.x;
    if (i < n) p[i] = v;
}

__global__ void k_deg(const long long* __restrict__ dst, float* __restrict__ deg, int E) {
    int e = blockIdx.x * blockDim.x + threadIdx.x;
    if (e < E) atomicAdd(&deg[(int)dst[e]], 1.0f);
}

__global__ void k_rsqrt(float* __restrict__ p, int n) {
    int i = blockIdx.x * blockDim.x + threadIdx.x;
    if (i < n) p[i] = rsqrtf(p[i]);
}

// ---------------------------------------------------------------------------
// Precision staging: f32 -> packed bf16 (coalesced, 8 elems/thread).
// ---------------------------------------------------------------------------
__global__ void k_f32_to_bf16(const float* __restrict__ in, __bf16* __restrict__ out, long n8) {
    long i = (long)blockIdx.x * blockDim.x + threadIdx.x;
    if (i >= n8) return;
    const float4* p = (const float4*)in + i * 2;
    float4 x0 = p[0], x1 = p[1];
    union { __bf16 h[8]; uint4 u; } r;
    r.h[0] = (__bf16)x0.x; r.h[1] = (__bf16)x0.y; r.h[2] = (__bf16)x0.z; r.h[3] = (__bf16)x0.w;
    r.h[4] = (__bf16)x1.x; r.h[5] = (__bf16)x1.y; r.h[6] = (__bf16)x1.z; r.h[7] = (__bf16)x1.w;
    ((uint4*)out)[i] = r.u;
}

// W[K][Nout] f32  ->  Wt[Nout][K] bf16  (tiny, one-time per layer)
__global__ void k_w_to_bf16_t(const float* __restrict__ W, __bf16* __restrict__ Wt,
                              int K, int Nout) {
    const int chunks = K >> 3;
    long i = (long)blockIdx.x * blockDim.x + threadIdx.x;
    if (i >= (long)Nout * chunks) return;
    const int col = (int)(i / chunks);
    const int k0  = (int)(i % chunks) << 3;
    union { __bf16 h[8]; uint4 u; } r;
#pragma unroll
    for (int j = 0; j < 8; ++j)
        r.h[j] = (__bf16)W[(long)(k0 + j) * Nout + col];
    ((uint4*)Wt)[i] = r.u;
}

// ---------------------------------------------------------------------------
// WMMA GEMM:  D[M,Nout] = A[M,K] @ W[K,Nout] (+ bias)
// A is packed bf16 row-major, W is packed bf16 transposed (Wt[Nout][K]).
// One wave32 -> 16 rows x 64 cols (4 n-tiles sharing the A fragment).
// Steady state per 32-K step: 10x global_load_b128 + 4x v_wmma_f32_16x16x32_bf16.
//
// Fragment layouts per CDNA5 ISA 7.12.2 (wave32):
//   A 16x32 : lane&15 = row ; elem j -> K = 8*(lane>>4)+j (j<8), +16 for j>=8
//   B 32x16 : lane&15 = col ; elem e -> K = 8*(lane>>4)+e (e<8), +16 for e>=8 (same as A)
//   C/D f32 : lane&15 = col ; VGPR r -> row = 8*(lane>>4)+r
// ---------------------------------------------------------------------------
template<int K, bool BIAS>
__global__ void k_wmma_gemm(const __bf16* __restrict__ A, const __bf16* __restrict__ Wt,
                            const float* __restrict__ bias, float* __restrict__ D,
                            int M, int Nout) {
    constexpr int NT = 4;                           // n-tiles per wave
    const int lane = threadIdx.x & 31;
    const int wid  = threadIdx.x >> 5;
    const int tm   = M >> 4;
    const int gm   = blockIdx.x * (blockDim.x >> 5) + wid;  // wave-uniform
    if (gm >= tm) return;                                   // uniform exit: EXEC all-ones at WMMA
    const int m0 = gm << 4;
    const int n0 = blockIdx.y * (NT * 16);

    const int hlane = lane & 15;
    const int hi    = lane >> 4;

    const __bf16* ap  = A  + (long)(m0 + hlane) * K + hi * 8;
    const __bf16* wp0 = Wt + (long)(n0 + hlane) * K + hi * 8;
    __builtin_prefetch(wp0, 0, 0);

    union Frag { uint4 u[2]; v16bf v; };
    v8f acc[NT] = {};

#pragma unroll
    for (int k0 = 0; k0 < K; k0 += 32) {
        Frag a;
        a.u[0] = *(const uint4*)(ap + k0);          // elems 0..7
        a.u[1] = *(const uint4*)(ap + k0 + 16);     // elems 8..15
#pragma unroll
        for (int t = 0; t < NT; ++t) {
            Frag b;
            const __bf16* wp = wp0 + (long)t * 16 * K + k0;
            b.u[0] = *(const uint4*)(wp);           // elems 0..7
            b.u[1] = *(const uint4*)(wp + 16);      // elems 8..15
            acc[t] = __builtin_amdgcn_wmma_f32_16x16x32_bf16(
                false, a.v, false, b.v, (short)0, acc[t], false, false);
        }
    }

    const int orow = m0 + hi * 8;
#pragma unroll
    for (int t = 0; t < NT; ++t) {
        const int ocol = n0 + t * 16 + hlane;
        const float bv = BIAS ? bias[ocol] : 0.0f;
#pragma unroll
        for (int r = 0; r < 8; ++r)                 // lane-coalesced 64B row segments
            D[(long)(orow + r) * Nout + ocol] = acc[t][r] + bv;
    }
}

// ---------------------------------------------------------------------------
// GCN aggregation:  agg = dinv^2 * h  (self loop)  then  agg[dst] += w_e*h[src]
// Accumulators are <=51 MB -> L2-resident float atomics (192 MB L2).
// ---------------------------------------------------------------------------
__global__ void k_agg_init(const float* __restrict__ h, const float* __restrict__ dinv,
                           float* __restrict__ agg, long total, int F) {
    long i = (long)blockIdx.x * blockDim.x + threadIdx.x;
    if (i >= total) return;
    float di = dinv[(int)(i / F)];
    agg[i] = h[i] * di * di;
}

__global__ void k_scatter(const float* __restrict__ h, float* __restrict__ agg,
                          const long long* __restrict__ src, const long long* __restrict__ dst,
                          const float* __restrict__ dinv, int E, int F) {
    const int  chunks = F >> 2;
    const long total  = (long)E * chunks;
    long i = (long)blockIdx.x * blockDim.x + threadIdx.x;
    if (i >= total) return;
    const int e = (int)(i / chunks);
    const int c = (int)(i % chunks) << 2;
    const int s = (int)src[e];
    const int d = (int)dst[e];
    const float w = dinv[s] * dinv[d];
    const float4 v = *(const float4*)(h + (long)s * F + c);
    float* out = agg + (long)d * F + c;
    atomicAdd(out + 0, v.x * w);
    atomicAdd(out + 1, v.y * w);
    atomicAdd(out + 2, v.z * w);
    atomicAdd(out + 3, v.w * w);
}

template<bool RELU>
__global__ void k_finalize(const float* __restrict__ agg, const float* __restrict__ bias,
                           float* __restrict__ out, long total, int F) {
    long i = (long)blockIdx.x * blockDim.x + threadIdx.x;
    if (i >= total) return;
    float v = agg[i] + bias[(int)(i % F)];
    if (RELU) v = fmaxf(v, 0.0f);
    out[i] = v;
}

// ---------------------------------------------------------------------------
// Link predictor: sigmoid([z[src] | z[dst]] @ lw + lb), one wave32 per edge.
// ---------------------------------------------------------------------------
__global__ void k_link(const float* __restrict__ z, const long long* __restrict__ src,
                       const long long* __restrict__ dst, const float* __restrict__ lw,
                       const float* __restrict__ lb, float* __restrict__ out, int E) {
    const int lane = threadIdx.x & 31;
    const int wid  = threadIdx.x >> 5;
    const long e   = (long)blockIdx.x * (blockDim.x >> 5) + wid;
    if (e >= E) return;
    const int s = (int)src[e];
    const int d = (int)dst[e];
    const float* zs = z + (long)s * 64;
    const float* zd = z + (long)d * 64;
    float acc = zs[lane] * lw[lane]      + zs[lane + 32] * lw[lane + 32]
              + zd[lane] * lw[lane + 64] + zd[lane + 32] * lw[lane + 96];
#pragma unroll
    for (int off = 16; off >= 1; off >>= 1)
        acc += __shfl_xor(acc, off, 32);
    if (lane == 0) out[e] = 1.0f / (1.0f + __expf(-(acc + lb[0])));
}

// ---------------------------------------------------------------------------
// Orchestration
// ---------------------------------------------------------------------------
static inline unsigned cdiv(long a, long b) { return (unsigned)((a + b - 1) / b); }

extern "C" void kernel_launch(void* const* d_in, const int* in_sizes, int n_in,
                              void* d_out, int out_size, void* d_ws, size_t ws_size,
                              hipStream_t stream) {
    const float*     x  = (const float*)d_in[0];
    const long long* ei = (const long long*)d_in[1];
    const float* ew1 = (const float*)d_in[2];
    const float* eb1 = (const float*)d_in[3];
    const float* ew2 = (const float*)d_in[4];
    const float* eb2 = (const float*)d_in[5];
    const float* efw = (const float*)d_in[6];
    const float* efb = (const float*)d_in[7];
    const float* dw1 = (const float*)d_in[8];
    const float* db1 = (const float*)d_in[9];
    const float* dw2 = (const float*)d_in[10];
    const float* db2 = (const float*)d_in[11];
    const float* dfw = (const float*)d_in[12];
    const float* dfb = (const float*)d_in[13];
    const float* lw  = (const float*)d_in[14];
    const float* lb  = (const float*)d_in[15];

    const int N = in_sizes[0] / 128;
    const int E = in_sizes[1] / 2;
    const long long* src = ei;
    const long long* dst = ei + E;

    // Workspace carve-out (256B aligned)
    char*  ws  = (char*)d_ws;
    size_t off = 0;
    auto carve = [&](size_t bytes) {
        void* p = ws + off;
        off = (off + bytes + 255) & ~(size_t)255;
        return p;
    };
    float*  dinv = (float*)carve((size_t)N * 4);
    float*  bufA = (float*)carve((size_t)N * 256 * 4);
    float*  bufB = (float*)carve((size_t)N * 256 * 4);
    float*  zbuf = (float*)carve((size_t)N * 64 * 4);
    __bf16* abf  = (__bf16*)carve((size_t)N * 256 * 2);   // bf16 staged activations
    __bf16* wtb  = (__bf16*)carve((size_t)128 * 1024 * 2); // bf16 transposed weights (max)

    float* xhat  = (float*)d_out;                     // [N,1024]
    float* probs = (float*)d_out + (size_t)N * 1024;  // [E]

    const int TB  = 256;
    const int WPB = TB / 32;

    auto gemm = [&](const float* A, const float* W, const float* bias, float* D,
                    int K, int Nout, bool biasf) {
        long n8 = (long)N * K / 8;
        k_f32_to_bf16<<<cdiv(n8, TB), TB, 0, stream>>>(A, abf, n8);
        long wn = (long)Nout * (K >> 3);
        k_w_to_bf16_t<<<cdiv(wn, TB), TB, 0, stream>>>(W, wtb, K, Nout);
        dim3 grid(cdiv(N >> 4, WPB), Nout / 64);
        if (K == 64) {
            if (biasf) k_wmma_gemm<64,  true ><<<grid, TB, 0, stream>>>(abf, wtb, bias, D, N, Nout);
            else       k_wmma_gemm<64,  false><<<grid, TB, 0, stream>>>(abf, wtb, bias, D, N, Nout);
        } else if (K == 128) {
            if (biasf) k_wmma_gemm<128, true ><<<grid, TB, 0, stream>>>(abf, wtb, bias, D, N, Nout);
            else       k_wmma_gemm<128, false><<<grid, TB, 0, stream>>>(abf, wtb, bias, D, N, Nout);
        } else {
            if (biasf) k_wmma_gemm<256, true ><<<grid, TB, 0, stream>>>(abf, wtb, bias, D, N, Nout);
            else       k_wmma_gemm<256, false><<<grid, TB, 0, stream>>>(abf, wtb, bias, D, N, Nout);
        }
    };
    auto gcn_agg = [&](const float* h, float* agg, float* dest, const float* bias,
                       int F, bool reluf) {
        long total = (long)N * F;
        k_agg_init<<<cdiv(total, TB), TB, 0, stream>>>(h, dinv, agg, total, F);
        long etotal = (long)E * (F >> 2);
        k_scatter<<<cdiv(etotal, TB), TB, 0, stream>>>(h, agg, src, dst, dinv, E, F);
        if (reluf) k_finalize<true ><<<cdiv(total, TB), TB, 0, stream>>>(agg, bias, dest, total, F);
        else       k_finalize<false><<<cdiv(total, TB), TB, 0, stream>>>(agg, bias, dest, total, F);
    };

    // degree -> dinv
    k_fill <<<cdiv(N, TB), TB, 0, stream>>>(dinv, 1.0f, N);   // self-loop
    k_deg  <<<cdiv(E, TB), TB, 0, stream>>>(dst, dinv, E);
    k_rsqrt<<<cdiv(N, TB), TB, 0, stream>>>(dinv, N);

    // Encoder: GCN(128->128)+ReLU, GCN(128->64), Linear(64->64)
    gemm(x,    ew1, nullptr, bufA, 128, 128, false);
    gcn_agg(bufA, bufB, bufA, eb1, 128, true);
    gemm(bufA, ew2, nullptr, bufB, 128, 64,  false);
    gcn_agg(bufB, bufA, bufB, eb2, 64,  false);
    gemm(bufB, efw, efb,     zbuf, 64,  64,  true);           // latent z [N,64]

    // Decoder: GCN(64->256)+ReLU, GCN(256->128), Linear(128->1024)
    gemm(zbuf, dw1, nullptr, bufA, 64,  256, false);
    gcn_agg(bufA, bufB, bufA, db1, 256, true);
    gemm(bufA, dw2, nullptr, bufB, 256, 128, false);
    gcn_agg(bufB, bufA, bufB, db2, 128, false);
    gemm(bufB, dfw, dfb,     xhat, 128, 1024, true);          // x_hat [N,1024]

    // Link predictor
    k_link<<<cdiv(E, WPB), TB, 0, stream>>>(zbuf, src, dst, lw, lb, probs, E);
}
